// PagedAttention_61959198212213
// MI455X (gfx1250) — matrix operand order repaired and checked by reading, not compiled, not run on hardware
//
#include <hip/hip_runtime.h>
#include <hip/hip_bf16.h>

#define HDIM   4096
#define NHEADS 32
#define HEADD  128
#define NBAT   8
#define NTOK   32
#define NPAST  4064
#define NKV    4096
#define NM     256            // NBAT*NTOK query rows
#define QK_SCALE 0.08838834764831845f   // 1/sqrt(128)

typedef __bf16 bf16_t;
typedef bf16_t v16bf __attribute__((ext_vector_type(16)));
typedef bf16_t v8bf  __attribute__((ext_vector_type(8)));
typedef float  v8f   __attribute__((ext_vector_type(8)));
typedef float  v4f   __attribute__((ext_vector_type(4)));

#define WMMA_BF16(a,b,c) \
  __builtin_amdgcn_wmma_f32_16x16x32_bf16(false,(a),false,(b),(short)0,(c),false,false)

// ---- CDNA5 async global->LDS copy (ASYNCcnt-tracked) ----------------------
// One instruction moves 32 lanes x 16B. LDS address = low 32 bits of the
// generic pointer (wave-relative LDS byte offset per ISA 10.2 aperture rules).
static __device__ __forceinline__ void async_b128(void* lds_dst, const void* gsrc) {
  uint32_t loff = (uint32_t)(size_t)lds_dst;
  asm volatile("global_load_async_to_lds_b128 %0, %1, off"
               :: "v"(loff), "v"(gsrc)
               : "memory");
}
#define WAIT_ASYNC_0()  asm volatile("s_wait_asynccnt 0x0"  ::: "memory")
#define WAIT_ASYNC_16() asm volatile("s_wait_asynccnt 0x10" ::: "memory")

// ---- fragment load helpers -------------------------------------------------
// A fragment (16x32, 16-bit): this lane needs K = [0..8) and [16..24) relative
// to its half-selected base (half8 already folded into the pointer).
static __device__ __forceinline__ v16bf ldA_f32(const float* p) {
  v4f a0 = *(const v4f*)(p);
  v4f a1 = *(const v4f*)(p + 4);
  v4f a2 = *(const v4f*)(p + 16);
  v4f a3 = *(const v4f*)(p + 20);
  v16bf r;
#pragma unroll
  for (int i = 0; i < 4; ++i) {
    r[i]      = (bf16_t)a0[i];
    r[4 + i]  = (bf16_t)a1[i];
    r[8 + i]  = (bf16_t)a2[i];
    r[12 + i] = (bf16_t)a3[i];
  }
  return r;
}
static __device__ __forceinline__ v16bf ldA_bf(const bf16_t* p) {
  v8bf lo = *(const v8bf*)(p);
  v8bf hi = *(const v8bf*)(p + 16);
  return __builtin_shufflevector(lo, hi, 0,1,2,3,4,5,6,7,8,9,10,11,12,13,14,15);
}
// B fragment row (lane = K-row): 16 contiguous N values.
static __device__ __forceinline__ v16bf ld16_f32(const float* p) {
  v16bf r;
#pragma unroll
  for (int i = 0; i < 16; i += 4) {
    v4f a = *(const v4f*)(p + i);
    r[i] = (bf16_t)a[0]; r[i+1] = (bf16_t)a[1];
    r[i+2] = (bf16_t)a[2]; r[i+3] = (bf16_t)a[3];
  }
  return r;
}
static __device__ __forceinline__ v16bf ld16_bf(const bf16_t* p) {
  v8bf lo = *(const v8bf*)(p);
  v8bf hi = *(const v8bf*)(p + 8);
  return __builtin_shufflevector(lo, hi, 0,1,2,3,4,5,6,7,8,9,10,11,12,13,14,15);
}

// ---- kernel 1: x[M,H] f32 -> xT[H,M] bf16 ---------------------------------
__global__ void k_transpose(const float* __restrict__ x, bf16_t* __restrict__ xT) {
  __shared__ float tile[32][33];
  int h0 = blockIdx.x * 32, m0 = blockIdx.y * 32;
  int tx = threadIdx.x, ty = threadIdx.y;
  for (int i = ty; i < 32; i += 8)
    tile[i][tx] = x[(size_t)(m0 + i) * HDIM + h0 + tx];
  __syncthreads();
  for (int i = ty; i < 32; i += 8)
    xT[(size_t)(h0 + i) * NM + m0 + tx] = (bf16_t)tile[tx][i];
}

// ---- kernel 2: fused QKV projection, y^T = W * x^T ------------------------
__global__ void k_qkv(const float* __restrict__ Wq, const float* __restrict__ Wk,
                      const float* __restrict__ Wv, const bf16_t* __restrict__ xT,
                      bf16_t* __restrict__ qb, bf16_t* __restrict__ kb,
                      bf16_t* __restrict__ vb) {
  int lane  = threadIdx.x & 31;
  int Wid   = blockIdx.x * 8 + (threadIdx.x >> 5);   // 0..1535
  int w     = Wid / 512;
  int r     = Wid - w * 512;
  int ntile = r >> 1;                                 // 0..255
  int mg    = r & 1;                                  // 0..1 (128 m each)
  int half8 = (lane >> 4) << 3;

  const float* Wp   = (w == 0) ? Wq : (w == 1) ? Wk : Wv;
  const float* arow = Wp + (size_t)(ntile * 16 + (lane & 15)) * HDIM + half8;

  v8f zero = {};
  v8f acc[8];
#pragma unroll
  for (int i = 0; i < 8; ++i) acc[i] = zero;

  for (int k0 = 0; k0 < HDIM; k0 += 32) {
    v16bf a = ldA_f32(arow + k0);                      // W rows stream (A)
    const bf16_t* bp = xT + (size_t)(k0 + half8) * NM + mg * 128 + (lane & 15);
#pragma unroll
    for (int nm = 0; nm < 8; ++nm) {
      v16bf b;
#pragma unroll
      for (int i = 0; i < 16; ++i) b[i] = bp[(size_t)(i + (i & 8)) * NM + nm * 16];
      acc[nm] = WMMA_BF16(a, b, acc[nm]);
    }
  }

  bf16_t* dst = (w == 0) ? qb : (w == 1) ? kb : vb;
  float sc = (w == 0) ? QK_SCALE : 1.f;
#pragma unroll
  for (int nm = 0; nm < 8; ++nm) {
#pragma unroll
    for (int j = 0; j < 8; ++j) {
      int n = ntile * 16 + j + half8;                  // hidden col
      int m = mg * 128 + nm * 16 + (lane & 15);        // query row
      int head = n >> 7, d = n & 127;
      int bb = m >> 5, t = m & 31;
      dst[(((size_t)bb * NHEADS + head) * NTOK + t) * HEADD + d] =
          (bf16_t)(acc[nm][j] * sc);
    }
  }
}

// ---- kernel 3: flash attention per (b,head) -------------------------------
// 8 waves = (qg in {0,1}) x (kw in {0..3}). The qg-pair with the same kw
// shares one async-staged K/V chunk in LDS (each chunk fetched ONCE).
// Pipeline per 32-kv chunk c:
//   issue V(c) async | S^T = K(c)*Q^T from LDS | barrier | issue K(c+4) async
//   | softmax | wait V + barrier | PV from LDS | wait K + barrier
__global__ void k_attn(const bf16_t* __restrict__ qb, const bf16_t* __restrict__ kb,
                       const bf16_t* __restrict__ vb, const float* __restrict__ pastK,
                       const float* __restrict__ pastV, bf16_t* __restrict__ attnT) {
  extern __shared__ float smem[];
  float* KS = smem;                         // 4 * 32*128 f32  (64 KB)
  float* VS = KS + 4 * 32 * HEADD;          // 4 * 32*128 f32  (64 KB)
  float* Oc = VS + 4 * 32 * HEADD;          // 32*128 f32      (16 KB)
  float* Mw = Oc + NTOK * HEADD;            // [2][4][16]
  float* Lw = Mw + 2 * 4 * 16;              // [2][4][16]

  int tid  = threadIdx.x;
  int lane = tid & 31;
  int wave = tid >> 5;
  int qg   = wave >> 2;
  int kw   = wave & 3;
  int b    = blockIdx.x >> 5;
  int head = blockIdx.x & 31;
  int half8 = (lane >> 4) << 3;
  const size_t bh = (size_t)b * NHEADS + head;

  for (int i = tid; i < NTOK * HEADD; i += 256) Oc[i] = 0.f;

  // Q^T B-fragments (lane%16 = m column, 16 hd values along the fragment), once.
  const bf16_t* qbase = qb + (bh * NTOK + qg * 16) * HEADD;
  v16bf QT[4];
#pragma unroll
  for (int ks = 0; ks < 4; ++ks) {
    int hd = ks * 32 + half8;
#pragma unroll
    for (int i = 0; i < 16; ++i) QT[ks][i] = qbase[(size_t)(lane & 15) * HEADD + hd + i + (i & 8)];
  }

  float* myKS = KS + kw * 32 * HEADD;
  float* myVS = VS + kw * 32 * HEADD;

  // prologue: stage my half (16 rows) of K(chunk kw)
  {
    const float* src = pastK + (bh * NPAST + (size_t)kw * 32 + qg * 16) * HEADD;
    float* dstl = myKS + (qg * 16) * HEADD;
#pragma unroll
    for (int i = 0; i < 16; ++i)
      async_b128(dstl + i * HEADD + lane * 4, src + (size_t)i * HEADD + lane * 4);
  }
  WAIT_ASYNC_0();
  __syncthreads();

  v8f zero = {};
  v8f O[8];
#pragma unroll
  for (int i = 0; i < 8; ++i) O[i] = zero;
  float mrow = -3.0e38f, lrow = 0.f;

  for (int c = kw; c < NKV / 32; c += 4) {
    int kvb = c * 32;
    bool isnew = (kvb >= NPAST);                        // chunk 127 only
    bool stageNext = (c + 4 < 127);

    // issue my half of V(c); overlaps with the QK^T WMMAs below
    if (!isnew) {
      const float* src = pastV + (bh * NPAST + (size_t)kvb + qg * 16) * HEADD;
      float* dstl = myVS + (qg * 16) * HEADD;
#pragma unroll
      for (int i = 0; i < 16; ++i)
        async_b128(dstl + i * HEADD + lane * 4, src + (size_t)i * HEADD + lane * 4);
    }

    // S^T = K_chunk * Q^T  (two 16-kv tiles, f32 accumulate over hd)
    v8f S0 = zero, S1 = zero;
#pragma unroll
    for (int ks = 0; ks < 4; ++ks) {
      v16bf a0, a1;
      if (!isnew) {
        const float* kr = myKS + (lane & 15) * HEADD + ks * 32 + half8;
        a0 = ldA_f32(kr);
        a1 = ldA_f32(kr + 16 * HEADD);
      } else {
        const bf16_t* kr = kb + (bh * NTOK + (kvb - NPAST) + (lane & 15)) * HEADD
                           + ks * 32 + half8;
        a0 = ldA_bf(kr);
        a1 = ldA_bf(kr + 16 * HEADD);
      }
      S0 = WMMA_BF16(a0, QT[ks], S0);
      S1 = WMMA_BF16(a1, QT[ks], S1);
    }
    __syncthreads();                 // qg-pair done reading K(c)

    // issue my half of K(c+4); overlaps with softmax + PV below
    if (stageNext) {
      const float* src = pastK + (bh * NPAST + (size_t)(kvb + 128) + qg * 16) * HEADD;
      float* dstl = myKS + (qg * 16) * HEADD;
#pragma unroll
      for (int i = 0; i < 16; ++i)
        async_b128(dstl + i * HEADD + lane * 4, src + (size_t)i * HEADD + lane * 4);
    }

    // online softmax (stats live per lane for row m%16 == lane%16)
    float cm = -3.0e38f;
#pragma unroll
    for (int j = 0; j < 8; ++j) cm = fmaxf(cm, fmaxf(S0[j], S1[j]));
    cm = fmaxf(cm, __shfl_xor(cm, 16, 32));
    float mnew = fmaxf(mrow, cm);
    float resc = __expf(mrow - mnew);
    float rs = 0.f;
    v16bf Pa;                                   // S^T tiles ARE the P A-frag
#pragma unroll
    for (int j = 0; j < 8; ++j) {
      float p0 = __expf(S0[j] - mnew);
      float p1 = __expf(S1[j] - mnew);
      rs += p0 + p1;
      Pa[j]     = (bf16_t)p0;
      Pa[8 + j] = (bf16_t)p1;
    }
    rs += __shfl_xor(rs, 16, 32);
    lrow = lrow * resc + rs;
    mrow = mnew;

    float sj[8];                                 // rescale, broadcast to O rows
#pragma unroll
    for (int j = 0; j < 8; ++j) sj[j] = __shfl(resc, j + half8, 32);

    // wait for V(c) (16 newer K(c+4) ops may stay in flight), sync the pair
    if (stageNext) { WAIT_ASYNC_16(); } else { WAIT_ASYNC_0(); }
    __syncthreads();

    // O += P * V  (V rows from LDS load directly as B fragments)
#pragma unroll
    for (int nt = 0; nt < 8; ++nt) {
      v16bf bv;
      if (!isnew) {
#pragma unroll
        for (int i = 0; i < 16; ++i)
          bv[i] = (bf16_t)myVS[(half8 + i + (i & 8)) * HEADD + nt * 16 + (lane & 15)];
      } else {
#pragma unroll
        for (int i = 0; i < 16; ++i)
          bv[i] = vb[(bh * NTOK + (kvb - NPAST) + half8 + i + (i & 8)) * HEADD + nt * 16 + (lane & 15)];
      }
      v8f o = O[nt];
#pragma unroll
      for (int j = 0; j < 8; ++j) o[j] *= sj[j];
      O[nt] = WMMA_BF16(Pa, bv, o);
    }

    // K(c+4) must be resident (and Vstage reads done) before next iteration
    WAIT_ASYNC_0();
    __syncthreads();
  }

  // cross-wave combine of the 4 kv partitions
  if (lane < 16) {
    Mw[(qg * 4 + kw) * 16 + lane] = mrow;
    Lw[(qg * 4 + kw) * 16 + lane] = lrow;
  }
  __syncthreads();
  float gm = -3.0e38f;
#pragma unroll
  for (int w2 = 0; w2 < 4; ++w2)
    gm = fmaxf(gm, Mw[(qg * 4 + w2) * 16 + (lane & 15)]);
  float gl = 0.f;
#pragma unroll
  for (int w2 = 0; w2 < 4; ++w2)
    gl += Lw[(qg * 4 + w2) * 16 + (lane & 15)] *
          __expf(Mw[(qg * 4 + w2) * 16 + (lane & 15)] - gm);
  float fs = __expf(mrow - gm) / gl;
  float fj[8];
#pragma unroll
  for (int j = 0; j < 8; ++j) fj[j] = __shfl(fs, j + half8, 32);
#pragma unroll
  for (int nt = 0; nt < 8; ++nt) {
#pragma unroll
    for (int j = 0; j < 8; ++j) {
      int mr = qg * 16 + j + half8;
      int d  = nt * 16 + (lane & 15);
      atomicAdd(&Oc[mr * HEADD + d], O[nt][j] * fj[j]);
    }
  }
  __syncthreads();

  // emit attn^T (bf16) so the out-projection streams Wo as A fragments
  for (int i = tid; i < NTOK * HEADD; i += 256) {
    int t = i >> 7, d = i & 127;
    attnT[((size_t)head * HEADD + d) * NM + (b * NTOK + t)] = (bf16_t)Oc[t * HEADD + d];
  }
}

// ---- kernel 4: out^T = Wo * attn^T, stored to d_out[m,n] ------------------
__global__ void k_oproj(const float* __restrict__ Wo, const bf16_t* __restrict__ attnT,
                        float* __restrict__ out) {
  int lane  = threadIdx.x & 31;
  int Wid   = blockIdx.x * 8 + (threadIdx.x >> 5);  // 0..511
  int ntile = Wid >> 1;
  int mg    = Wid & 1;
  int half8 = (lane >> 4) << 3;
  const float* arow = Wo + (size_t)(ntile * 16 + (lane & 15)) * HDIM + half8;

  v8f zero = {};
  v8f acc[8];
#pragma unroll
  for (int i = 0; i < 8; ++i) acc[i] = zero;

  for (int k0 = 0; k0 < HDIM; k0 += 32) {
    v16bf a = ldA_f32(arow + k0);
    const bf16_t* bp = attnT + (size_t)(k0 + half8) * NM + mg * 128 + (lane & 15);
#pragma unroll
    for (int nm = 0; nm < 8; ++nm) {
      v16bf b;
#pragma unroll
      for (int i = 0; i < 16; ++i) b[i] = bp[(size_t)(i + (i & 8)) * NM + nm * 16];
      acc[nm] = WMMA_BF16(a, b, acc[nm]);
    }
  }
#pragma unroll
  for (int nm = 0; nm < 8; ++nm) {
#pragma unroll
    for (int j = 0; j < 8; ++j) {
      int n = ntile * 16 + j + half8;
      int m = mg * 128 + nm * 16 + (lane & 15);
      out[(size_t)m * HDIM + n] = acc[nm][j];
    }
  }
}

// ---- host launcher ---------------------------------------------------------
extern "C" void kernel_launch(void* const* d_in, const int* in_sizes, int n_in,
                              void* d_out, int out_size, void* d_ws, size_t ws_size,
                              hipStream_t stream) {
  (void)in_sizes; (void)n_in; (void)out_size; (void)ws_size;
  const float* x     = (const float*)d_in[0];
  const float* pastK = (const float*)d_in[1];
  const float* pastV = (const float*)d_in[2];
  const float* Wq    = (const float*)d_in[3];
  const float* Wk    = (const float*)d_in[4];
  const float* Wv    = (const float*)d_in[5];
  const float* Wo    = (const float*)d_in[6];
  float* out = (float*)d_out;

  // workspace: 5 x (4096*256) bf16 buffers = 10 MB
  bf16_t* xT    = (bf16_t*)d_ws;
  bf16_t* qb    = xT   + (size_t)HDIM * NM;
  bf16_t* kbuf  = qb   + (size_t)HDIM * NM;
  bf16_t* vbuf  = kbuf + (size_t)HDIM * NM;
  bf16_t* attnT = vbuf + (size_t)HDIM * NM;

  // dynamic LDS for k_attn: K-stage + V-stage + combine buffers (~145 KB)
  size_t attn_lds = (size_t)(4 * 32 * HEADD + 4 * 32 * HEADD + NTOK * HEADD +
                             2 * 4 * 16 + 2 * 4 * 16) * sizeof(float);

  k_transpose<<<dim3(HDIM / 32, NM / 32), dim3(32, 8), 0, stream>>>(x, xT);
  k_qkv<<<192, 256, 0, stream>>>(Wq, Wk, Wv, xT, qb, kbuf, vbuf);
  k_attn<<<NBAT * NHEADS, 256, attn_lds, stream>>>(qb, kbuf, vbuf, pastK, pastV, attnT);
  k_oproj<<<64, 256, 0, stream>>>(Wo, attnT, out);
}
